// TM_mixing_time_60327110639927
// MI455X (gfx1250) — matrix operand order repaired and checked by reading, not compiled, not run on hardware
//
#include <hip/hip_runtime.h>

typedef _Float16 half_t;
typedef __attribute__((ext_vector_type(16))) _Float16 v16h;
typedef __attribute__((ext_vector_type(8)))  float    v8f;

#define INV_TAU 0.5714285714285714f   // 1/1.75
#define S3C     0.3535533905932738f   // (1/sqrt(2))^3

// ---------------------------------------------------------------------------
// Stage 1: LIF over T on input x (T,B,C,H,W) -> spikes s0 in (B,C,T,H,W)
// ---------------------------------------------------------------------------
__global__ void k_lif_input(const float* __restrict__ x, float* __restrict__ s0) {
  int gid = blockIdx.x * blockDim.x + threadIdx.x;
  if (gid >= 16 * 256 * 1024) return;
  int hw = gid & 1023;
  int bc = gid >> 10;                       // b*256 + c
  float v = 0.f;
  #pragma unroll
  for (int t = 0; t < 8; ++t) {
    float xt = x[((size_t)t * 4096 + bc) * 1024 + hw];
    float h  = v + (xt - v) * INV_TAU;
    float sp = (h >= 0.75f) ? 1.f : 0.f;
    v = h * (1.f - sp);
    s0[((size_t)bc * 8 + t) * 1024 + hw] = sp;
  }
}

// ---------------------------------------------------------------------------
// 3-D Haar DWT level 1: s0 (B,C,8,32,32) -> lll1 (B,C,4,16,16)
//                       + high1 raw in layout (b,j,hh,ww,c,t)  j=0..6, t=0..3
// ---------------------------------------------------------------------------
__global__ void k_dwt1(const float* __restrict__ s0, float* __restrict__ lll1,
                       float* __restrict__ h1raw) {
  int gid = blockIdx.x * blockDim.x + threadIdx.x;
  if (gid >= 16 * 256 * 1024) return;
  int wi = gid & 15; int r = gid >> 4;
  int hi = r & 15;   r >>= 4;
  int ti = r & 3;    int bc = r >> 2;
  const float* base = s0 + (size_t)bc * 8192 + (size_t)(2 * ti) * 1024 + (2 * hi) * 32 + 2 * wi;
  float xv[8];
  #pragma unroll
  for (int a = 0; a < 2; ++a)
    for (int b2 = 0; b2 < 2; ++b2)
      for (int c2 = 0; c2 < 2; ++c2)
        xv[a * 4 + b2 * 2 + c2] = base[a * 1024 + b2 * 32 + c2];
  int b = bc >> 8, c = bc & 255;
  #pragma unroll
  for (int j = 0; j < 8; ++j) {
    float acc = 0.f;
    #pragma unroll
    for (int a = 0; a < 2; ++a)
      for (int b2 = 0; b2 < 2; ++b2)
        for (int c2 = 0; c2 < 2; ++c2) {
          float sg = 1.f;
          if ((j & 4) && a)  sg = -sg;
          if ((j & 2) && b2) sg = -sg;
          if ((j & 1) && c2) sg = -sg;
          acc += sg * xv[a * 4 + b2 * 2 + c2];
        }
    acc *= S3C;
    if (j == 0)
      lll1[(size_t)bc * 1024 + ti * 256 + hi * 16 + wi] = acc;
    else
      h1raw[((size_t)(((b * 7 + (j - 1)) * 16 + hi) * 16 + wi) * 256 + c) * 4 + ti] = acc;
  }
}

// Level 2: lll1 (B,C,4,16,16) -> x_lll (B,C,2,8,8) + high2 raw (b,j,hh,ww,c,t) t=0..1
__global__ void k_dwt2(const float* __restrict__ lll1, float* __restrict__ xlll,
                       float* __restrict__ h2raw) {
  int gid = blockIdx.x * blockDim.x + threadIdx.x;
  if (gid >= 16 * 256 * 128) return;
  int wi = gid & 7; int r = gid >> 3;
  int hi = r & 7;   r >>= 3;
  int ti = r & 1;   int bc = r >> 1;
  const float* base = lll1 + (size_t)bc * 1024 + (2 * ti) * 256 + (2 * hi) * 16 + 2 * wi;
  float xv[8];
  #pragma unroll
  for (int a = 0; a < 2; ++a)
    for (int b2 = 0; b2 < 2; ++b2)
      for (int c2 = 0; c2 < 2; ++c2)
        xv[a * 4 + b2 * 2 + c2] = base[a * 256 + b2 * 16 + c2];
  int b = bc >> 8, c = bc & 255;
  #pragma unroll
  for (int j = 0; j < 8; ++j) {
    float acc = 0.f;
    #pragma unroll
    for (int a = 0; a < 2; ++a)
      for (int b2 = 0; b2 < 2; ++b2)
        for (int c2 = 0; c2 < 2; ++c2) {
          float sg = 1.f;
          if ((j & 4) && a)  sg = -sg;
          if ((j & 2) && b2) sg = -sg;
          if ((j & 1) && c2) sg = -sg;
          acc += sg * xv[a * 4 + b2 * 2 + c2];
        }
    acc *= S3C;
    if (j == 0)
      xlll[(size_t)bc * 128 + ti * 64 + hi * 8 + wi] = acc;
    else
      h2raw[((size_t)(((b * 7 + (j - 1)) * 8 + hi) * 8 + wi) * 256 + c) * 2 + ti] = acc;
  }
}

// ---------------------------------------------------------------------------
// x_lll path: LIF(0.55) -> *w_ll (C,2,8,8) -> LIF(0.55)
// ---------------------------------------------------------------------------
__global__ void k_lll_path(const float* __restrict__ lll, const float* __restrict__ w_ll,
                           float* __restrict__ out) {
  int gid = blockIdx.x * blockDim.x + threadIdx.x;
  if (gid >= 16 * 256 * 64) return;
  int wi = gid & 7; int r = gid >> 3;
  int hi = r & 7;   int bc = r >> 3;
  int c = bc & 255;
  float v1 = 0.f, v2 = 0.f;
  #pragma unroll
  for (int t = 0; t < 2; ++t) {
    size_t idx = (size_t)bc * 128 + t * 64 + hi * 8 + wi;
    float xt = lll[idx];
    float h1 = v1 + (xt - v1) * INV_TAU;
    float s1 = (h1 >= 0.55f) ? 1.f : 0.f;
    v1 = h1 * (1.f - s1);
    float y = s1 * w_ll[((size_t)(c * 2 + t) * 8 + hi) * 8 + wi];
    float h2 = v2 + (y - v2) * INV_TAU;
    float s2 = (h2 >= 0.55f) ? 1.f : 0.f;
    v2 = h2 * (1.f - s2);
    out[idx] = s2;
  }
}

// ---------------------------------------------------------------------------
// Branch stage kernels; working layout is an (M,256) row-major matrix,
// rows r = t*P + pos, pos = ((b*7+j)*HH+hh)*WW+ww
// ---------------------------------------------------------------------------
__global__ void k_lif_wA(const float* __restrict__ raw, const float* __restrict__ wA,
                         float* __restrict__ u, int P, int tlen, int HH, int WW) {
  int c = threadIdx.x;
  int pos = blockIdx.x;
  int ww = pos % WW; int r = pos / WW;
  int hh = r % HH;   r /= HH;
  int j  = r % 7;
  const float* rp = raw + ((size_t)pos * 256 + c) * tlen;
  const float* wp = wA + ((size_t)(j * HH + hh) * WW + ww) * tlen;
  float v = 0.f;
  for (int t = 0; t < tlen; ++t) {
    float xt = rp[t];
    float h  = v + (xt - v) * INV_TAU;
    float sp = (h >= 0.75f) ? 1.f : 0.f;
    v = h * (1.f - sp);
    u[((size_t)t * P + pos) * 256 + c] = sp * wp[t];
  }
}

__global__ void k_bn_lif_wB(const float* __restrict__ u, const float* __restrict__ ss,
                            const float* __restrict__ wB, float* __restrict__ u2,
                            int P, int tlen, int HH, int WW) {
  int c = threadIdx.x;
  int pos = blockIdx.x;
  int ww = pos % WW; int r = pos / WW;
  int hh = r % HH;   r /= HH;
  int j  = r % 7;
  const float* wp = wB + ((size_t)(j * HH + hh) * WW + ww) * tlen;
  float scale = ss[c], shift = ss[256 + c];
  float v = 0.f;
  for (int t = 0; t < tlen; ++t) {
    size_t idx = ((size_t)t * P + pos) * 256 + c;
    float y = u[idx] * scale + shift;
    float h = v + (y - v) * INV_TAU;
    float sp = (h >= 0.75f) ? 1.f : 0.f;
    v = h * (1.f - sp);
    u2[idx] = sp * wp[t];
  }
}

template <typename OutT>
__global__ void k_bn_lif_out(const float* __restrict__ y, const float* __restrict__ ss,
                             OutT* __restrict__ out, int P, int tlen) {
  int c = threadIdx.x;
  int pos = blockIdx.x;
  float scale = ss[c], shift = ss[256 + c];
  float v = 0.f;
  for (int t = 0; t < tlen; ++t) {
    size_t idx = ((size_t)t * P + pos) * 256 + c;
    float yy = y[idx] * scale + shift;
    float h  = v + (yy - v) * INV_TAU;
    float sp = (h >= 0.75f) ? 1.f : 0.f;
    v = h * (1.f - sp);
    out[idx] = (OutT)sp;
  }
}

// ---------------------------------------------------------------------------
// BatchNorm statistics: deterministic two-stage per-channel reduction.
// ---------------------------------------------------------------------------
__global__ void k_bn_stats(const float* __restrict__ X, float* __restrict__ part, int M) {
  int c = threadIdx.x;                        // channel
  int r0 = blockIdx.x * 256;
  int rend = r0 + 256; if (rend > M) rend = M;
  float s = 0.f, q = 0.f;
  for (int r = r0; r < rend; ++r) {
    float v = X[(size_t)r * 256 + c];
    s += v; q += v * v;
  }
  part[(size_t)blockIdx.x * 512 + c]       = s;
  part[(size_t)blockIdx.x * 512 + 256 + c] = q;
}

__global__ void k_bn_reduce(const float* __restrict__ part, float* __restrict__ stats, int nb) {
  int t = threadIdx.x;                        // 0..511
  float s = 0.f;
  for (int i = 0; i < nb; ++i) s += part[(size_t)i * 512 + t];
  stats[t] = s;                               // [0..255]=sum, [256..511]=sumsq
}

__global__ void k_bn_finalize(float* __restrict__ stats, const float* __restrict__ g,
                              const float* __restrict__ b, float invM) {
  int c = threadIdx.x;
  float mean = stats[c] * invM;
  float var  = stats[256 + c] * invM - mean * mean;
  float sc   = g[c] * rsqrtf(var + 1e-5f);
  stats[512 + c] = sc;                        // scale
  stats[768 + c] = b[c] - mean * sc;          // shift
}

// ---------------------------------------------------------------------------
// Block-diagonal channel mix via WMMA:
//   Y[r, blk*32+k] = sum_d A[r, blk*32+d] * W[blk][d][k] + bias[blk][k]
// A: (M,256) f16 spikes, W: (8,32,32) f32, bias: (8,32) f32, Y: (M,256) f32.
// One wave = one 16x32 output tile (2 x v_wmma_f32_16x16x32_f16, K=32).
// ---------------------------------------------------------------------------
__global__ __launch_bounds__(256) void k_mix_wmma(const half_t* __restrict__ A,
                                                  const float* __restrict__ W,
                                                  const float* __restrict__ bias,
                                                  float* __restrict__ Y, int M) {
  int lane = threadIdx.x & 31;
  int wave = threadIdx.x >> 5;                // 8 waves / block
  int blk  = blockIdx.y;                      // channel block 0..7
  int rowTile = blockIdx.x * 8 + wave;        // 16-row tile
  if (rowTile * 16 >= M) return;              // never triggers (M % 128 == 0)
  int m0 = rowTile * 16;

  int n    = lane & 15;
  int kofs = (lane >= 16) ? 16 : 0;

  // B fragments: VGPR j holds K = 2j+kofs, 2j+1+kofs for column N = lane&15
  const float* Wb = W + (size_t)blk * 1024;
  v16h bf0, bf1;
  #pragma unroll
  for (int j = 0; j < 8; ++j) {
    int k = 2 * j + kofs;
    bf0[2 * j]     = (half_t)Wb[k * 32 + n];
    bf0[2 * j + 1] = (half_t)Wb[(k + 1) * 32 + n];
    bf1[2 * j]     = (half_t)Wb[k * 32 + 16 + n];
    bf1[2 * j + 1] = (half_t)Wb[(k + 1) * 32 + 16 + n];
  }

  // A fragment: row M = lane&15; K = 16*(lane>=16) + element (contiguous, matches B)
  const half_t* Arow = A + (size_t)(m0 + n) * 256 + blk * 32;
  int khalf = (lane >= 16) ? 8 : 0;
  v16h af;
  #pragma unroll
  for (int i = 0; i < 8; ++i) {
    int k = 2 * i + 2 * khalf;
    af[2 * i]     = Arow[k];
    af[2 * i + 1] = Arow[k + 1];
  }

  v8f c0 = {}, c1 = {};
  c0 = __builtin_amdgcn_wmma_f32_16x16x32_f16(false, af, false, bf0, (short)0, c0, false, false);
  c1 = __builtin_amdgcn_wmma_f32_16x16x32_f16(false, af, false, bf1, (short)0, c1, false, false);

  float bias0 = bias[blk * 32 + n];
  float bias1 = bias[blk * 32 + 16 + n];
  int mrow = (lane >= 16) ? 8 : 0;
  float* Yb = Y + (size_t)m0 * 256 + blk * 32;
  #pragma unroll
  for (int r = 0; r < 8; ++r) {
    Yb[(size_t)(mrow + r) * 256 + n]      = c0[r] + bias0;
    Yb[(size_t)(mrow + r) * 256 + 16 + n] = c1[r] + bias1;
  }
}

// ---------------------------------------------------------------------------
// Inverse DWT level 2: lllp (B,C,2,8,8) + f2 (t,b,j,8,8,c) -> ll (B,C,4,16,16)
// ---------------------------------------------------------------------------
__global__ void k_idwt2(const float* __restrict__ lllp, const float* __restrict__ f2,
                        float* __restrict__ ll) {
  int gid = blockIdx.x * blockDim.x + threadIdx.x;
  if (gid >= 16 * 256 * 128) return;
  int wi = gid & 7; int r = gid >> 3;
  int hi = r & 7;   r >>= 3;
  int ti = r & 1;   int bc = r >> 1;
  int b = bc >> 8, c = bc & 255;
  float band[8];
  band[0] = lllp[(size_t)bc * 128 + ti * 64 + hi * 8 + wi];
  #pragma unroll
  for (int jm = 0; jm < 7; ++jm)
    band[jm + 1] = f2[((size_t)(((ti * 16 + b) * 7 + jm) * 8 + hi) * 8 + wi) * 256 + c];
  float* o = ll + (size_t)bc * 1024 + (2 * ti) * 256 + (2 * hi) * 16 + 2 * wi;
  #pragma unroll
  for (int a = 0; a < 2; ++a)
    for (int b2 = 0; b2 < 2; ++b2)
      for (int c2 = 0; c2 < 2; ++c2) {
        float acc = 0.f;
        #pragma unroll
        for (int j = 0; j < 8; ++j) {
          float sg = 1.f;
          if ((j & 4) && a)  sg = -sg;
          if ((j & 2) && b2) sg = -sg;
          if ((j & 1) && c2) sg = -sg;
          acc += sg * band[j];
        }
        o[a * 256 + b2 * 16 + c2] = acc * S3C;
      }
}

// Inverse DWT level 1: ll (B,C,4,16,16) + f1 (t,b,j,16,16,c) -> out (B,C,8,32,32)
__global__ void k_idwt1(const float* __restrict__ ll, const float* __restrict__ f1,
                        float* __restrict__ out) {
  int gid = blockIdx.x * blockDim.x + threadIdx.x;
  if (gid >= 16 * 256 * 1024) return;
  int wi = gid & 15; int r = gid >> 4;
  int hi = r & 15;   r >>= 4;
  int ti = r & 3;    int bc = r >> 2;
  int b = bc >> 8, c = bc & 255;
  float band[8];
  band[0] = ll[(size_t)bc * 1024 + ti * 256 + hi * 16 + wi];
  #pragma unroll
  for (int jm = 0; jm < 7; ++jm)
    band[jm + 1] = f1[((size_t)(((ti * 16 + b) * 7 + jm) * 16 + hi) * 16 + wi) * 256 + c];
  float* o = out + (size_t)bc * 8192 + (size_t)(2 * ti) * 1024 + (2 * hi) * 32 + 2 * wi;
  #pragma unroll
  for (int a = 0; a < 2; ++a)
    for (int b2 = 0; b2 < 2; ++b2)
      for (int c2 = 0; c2 < 2; ++c2) {
        float acc = 0.f;
        #pragma unroll
        for (int j = 0; j < 8; ++j) {
          float sg = 1.f;
          if ((j & 4) && a)  sg = -sg;
          if ((j & 2) && b2) sg = -sg;
          if ((j & 1) && c2) sg = -sg;
          acc += sg * band[j];
        }
        o[a * 1024 + b2 * 32 + c2] = acc * S3C;
      }
}

// ---------------------------------------------------------------------------
// Host launcher
// ---------------------------------------------------------------------------
extern "C" void kernel_launch(void* const* d_in, const int* in_sizes, int n_in,
                              void* d_out, int out_size, void* d_ws, size_t ws_size,
                              hipStream_t stream) {
  (void)in_sizes; (void)n_in; (void)out_size; (void)ws_size;

  const float* x     = (const float*)d_in[0];
  const float* w_ll  = (const float*)d_in[1];
  const float* w1    = (const float*)d_in[2];
  const float* w12   = (const float*)d_in[3];
  const float* w2    = (const float*)d_in[4];
  const float* w22   = (const float*)d_in[5];
  const float* b1    = (const float*)d_in[6];
  const float* b12   = (const float*)d_in[7];
  const float* b2    = (const float*)d_in[8];
  const float* b22   = (const float*)d_in[9];
  const float* wlh1  = (const float*)d_in[10];
  const float* wlh12 = (const float*)d_in[11];
  const float* wlh2  = (const float*)d_in[12];
  const float* wlh22 = (const float*)d_in[13];
  const float* bn1_g  = (const float*)d_in[14];
  const float* bn1_b  = (const float*)d_in[15];
  const float* bn12_g = (const float*)d_in[16];
  const float* bn12_b = (const float*)d_in[17];
  const float* bn2_g  = (const float*)d_in[18];
  const float* bn2_b  = (const float*)d_in[19];
  const float* bn22_g = (const float*)d_in[20];
  const float* bn22_b = (const float*)d_in[21];
  const float* bn3_g  = (const float*)d_in[22];
  const float* bn3_b  = (const float*)d_in[23];
  const float* bn31_g = (const float*)d_in[24];
  const float* bn31_b = (const float*)d_in[25];
  const float* bn4_g  = (const float*)d_in[26];
  const float* bn4_b  = (const float*)d_in[27];
  const float* bn41_g = (const float*)d_in[28];
  const float* bn41_b = (const float*)d_in[29];

  float* ws = (float*)d_ws;
  // workspace layout (float offsets); W3 aliases W0 (s0 dead after DWT1)
  const size_t OFF_W0 = 0;            // 33,554,432  s0 / s2(h1,f16) / y2(h1)
  const size_t OFF_W1 = 33554432;     // 29,360,128  h1raw / u2 / s3(f16)
  const size_t OFF_W2 = 62914560;     // 29,360,128  u / y1 / f1
  const size_t OFF_V0 = 92274688;     //  3,670,016  h2raw / u2 / s3(f16)
  const size_t OFF_V1 = 95944704;     //  3,670,016  u / y1 / f2
  const size_t OFF_V2 = 99614720;     //  3,670,016  s2(f16) / y2
  const size_t OFF_L1 = 103284736;    //  4,194,304  lll1 / ll
  const size_t OFF_L2 = 107479040;    //    524,288  x_lll
  const size_t OFF_L3 = 108003328;    //    524,288  lllp
  const size_t OFF_ST = 108527616;    //      1,024  sums/sumsq/scale/shift
  const size_t OFF_PT = 108528640;    //    229,376  stat partials (448*512)

  float* stats = ws + OFF_ST;
  float* part  = ws + OFF_PT;
  const float* ss = stats + 512;      // scale/shift view

  auto run_bn = [&](const float* X, int M, const float* g, const float* bb) {
    int nb = (M + 255) / 256;
    k_bn_stats<<<nb, 256, 0, stream>>>(X, part, M);
    k_bn_reduce<<<1, 512, 0, stream>>>(part, stats, nb);
    k_bn_finalize<<<1, 256, 0, stream>>>(stats, g, bb, 1.0f / (float)M);
  };

  auto run_branch = [&](const float* raw, float* u, float* u2, half_t* s2, float* y1,
                        half_t* s3, float* y2, float* fout,
                        int P, int tlen, int HH, int WW,
                        const float* wA, const float* gA, const float* bA,
                        const float* wB, const float* gB, const float* bB,
                        const float* Wm1, const float* bm1, const float* gC, const float* bC,
                        const float* Wm2, const float* bm2, const float* gD, const float* bD) {
    int M = P * tlen;
    dim3 gg((unsigned)(M / 128), 8);
    k_lif_wA<<<P, 256, 0, stream>>>(raw, wA, u, P, tlen, HH, WW);
    run_bn(u, M, gA, bA);
    k_bn_lif_wB<<<P, 256, 0, stream>>>(u, ss, wB, u2, P, tlen, HH, WW);
    run_bn(u2, M, gB, bB);
    k_bn_lif_out<half_t><<<P, 256, 0, stream>>>(u2, ss, s2, P, tlen);
    k_mix_wmma<<<gg, 256, 0, stream>>>(s2, Wm1, bm1, y1, M);
    run_bn(y1, M, gC, bC);
    k_bn_lif_out<half_t><<<P, 256, 0, stream>>>(y1, ss, s3, P, tlen);
    k_mix_wmma<<<gg, 256, 0, stream>>>(s3, Wm2, bm2, y2, M);
    run_bn(y2, M, gD, bD);
    k_bn_lif_out<float><<<P, 256, 0, stream>>>(y2, ss, fout, P, tlen);
  };

  // 1) input LIF  2) DWT level 1  3) DWT level 2  4) lll path
  k_lif_input<<<16384, 256, 0, stream>>>(x, ws + OFF_W0);
  k_dwt1<<<16384, 256, 0, stream>>>(ws + OFF_W0, ws + OFF_L1, ws + OFF_W1);
  k_dwt2<<<2048, 256, 0, stream>>>(ws + OFF_L1, ws + OFF_L2, ws + OFF_V0);
  k_lll_path<<<1024, 256, 0, stream>>>(ws + OFF_L2, w_ll, ws + OFF_L3);

  // 5) high1 branch: P = 16*7*16*16 = 28672, t = 4
  run_branch(ws + OFF_W1, ws + OFF_W2, ws + OFF_W1, (half_t*)(ws + OFF_W0), ws + OFF_W2,
             (half_t*)(ws + OFF_W1), ws + OFF_W0, ws + OFF_W2,
             28672, 4, 16, 16,
             wlh1, bn3_g, bn3_b, wlh12, bn31_g, bn31_b,
             w1, b1, bn1_g, bn1_b, w12, b12, bn12_g, bn12_b);

  // 6) high2 branch: P = 16*7*8*8 = 7168, t = 2
  run_branch(ws + OFF_V0, ws + OFF_V1, ws + OFF_V0, (half_t*)(ws + OFF_V2), ws + OFF_V1,
             (half_t*)(ws + OFF_V0), ws + OFF_V2, ws + OFF_V1,
             7168, 2, 8, 8,
             wlh2, bn4_g, bn4_b, wlh22, bn41_g, bn41_b,
             w2, b2, bn2_g, bn2_b, w22, b22, bn22_g, bn22_b);

  // 7) inverse DWT: level 2 then level 1 straight into d_out ((B,C,T,H,W) order
  //    == the reference's raw reshape to (T,B,C,H,W))
  k_idwt2<<<2048, 256, 0, stream>>>(ws + OFF_L3, ws + OFF_V1, ws + OFF_L1);
  k_idwt1<<<16384, 256, 0, stream>>>(ws + OFF_L1, ws + OFF_W2, (float*)d_out);
}
